// YOLOv5_37718402794049
// MI455X (gfx1250) — matrix operand order repaired and checked by reading, not compiled, not run on hardware
//
#include <hip/hip_runtime.h>

// ---------------- CDNA5 WMMA types ----------------
typedef __attribute__((ext_vector_type(16))) __bf16 v16bf;
typedef __attribute__((ext_vector_type(8)))  float  v8f;

#define NUM_CLASSES 80
#define L 85
#define NA 3
#define TOTAL_ANCHORS 25200
#define SCORE_T 0.1f
#define NMS_T 0.6f
#define DETECTIONS 100
#define TOPK 1000
#define TOPK_PAD 1024
#define CAND_CAP 4096
#define IMG 640.0f
#define NIMG 8

// fast sigmoid: v_exp_f32 + v_rcp_f32 (reference sigmoid is approximate anyway)
__device__ __forceinline__ float sigm(float x) {
    return __builtin_amdgcn_rcpf(1.0f + __expf(-x));
}

// ---------------- Kernel 0: init counters ----------------
__global__ void init_kernel(int* candCnt) {
    if (threadIdx.x < NIMG) candCnt[threadIdx.x] = 0;
}

// ---- Kernel 0b: f32 -> bf16 weight pre-conversion, padded to 256 rows -------
__global__ void convert_weights_kernel(const float* __restrict__ src,
                                       __bf16* __restrict__ dst, int cin) {
    int i = blockIdx.x * blockDim.x + threadIdx.x;
    if (i < 256 * cin) {
        int mrow = i / cin;
        dst[i] = (mrow < 255) ? (__bf16)src[i] : (__bf16)0.0f;
    }
}

// ---- Kernel 0c: pad bias 255 -> 256 ------------------------------------------
__global__ void pad_bias_kernel(const float* __restrict__ src, float* __restrict__ dst) {
    int i = threadIdx.x;
    dst[i] = (i < 255) ? src[i] : 0.0f;
}

// ---------------- Kernel 1: WMMA GEMM + sigmoid + decode + candidate select ----
// block: 512 threads (16 waves). grid: (HW/16, N).
// Wave w computes out-channel rows [16w,16w+16) x 16 positions.
// Activations staged once per block into LDS as bf16 (one conversion/element);
// weights pre-converted+padded in global (L2 resident). A/B fragments are
// double-buffered so next chunk's loads overlap the current WMMA.
template <int CIN>
__global__ __launch_bounds__(512) void decode_level_kernel(
    const float* __restrict__ feat, const __bf16* __restrict__ wbf,
    const float* __restrict__ biasPad, float* __restrict__ boxesOut,
    float* __restrict__ candScore, int* __restrict__ candIdx, int* __restrict__ candCnt,
    int H, int W, float stride,
    float ax0, float ay0, float ax1, float ay1, float ax2, float ay2,
    int levelBase)
{
    constexpr int CINP = CIN + 16;                 // row pad: avoids LDS bank collisions
    __shared__ __align__(16) __bf16 sF[16][CINP];  // [pos][k] activation tile (bf16)
    __shared__ float sP[16][256];                  // [pos][out-chan] sigmoid outputs

    const int HW = H * W;
    const int n  = blockIdx.y;
    const int hwBase = blockIdx.x * 16;

    const int tid  = threadIdx.x;
    const int wave = tid >> 5;
    const int lane = tid & 31;
    const int half = lane >> 4;          // 0: lanes 0-15, 1: lanes 16-31
    const int l15  = lane & 15;

    // ---- stage activation tile: f32 global -> bf16 LDS (one conversion/element)
    {
        const float* fbase = feat + (size_t)n * CIN * HW + hwBase;
        for (int e = tid; e < CIN * 4; e += 512) {
            int pos = e & 15;
            int kq  = (e >> 4) * 4;
#pragma unroll
            for (int j = 0; j < 4; ++j)
                sF[pos][kq + j] = (__bf16)fbase[(size_t)(kq + j) * HW + pos];
        }
    }
    __syncthreads();

    // ---- GEMM: acc[16x16] = W[16 x CIN] * F[CIN x 16] ----
    const int m = wave * 16 + l15;       // A-matrix row for this lane (0..255)
    const __bf16* wr  = wbf + (size_t)m * CIN + half * 8;   // + kc per chunk
    const __bf16* bp0 = &sF[l15][half * 8];                 // + kc per chunk

    union Frag { uint4 q[2]; v16bf v; };
    Frag A[2], B[2];

    // preload chunk 0
    A[0].q[0] = *(const uint4*)(wr);
    A[0].q[1] = *(const uint4*)(wr + 16);
    B[0].q[0] = *(const uint4*)(bp0);
    B[0].q[1] = *(const uint4*)(bp0 + 16);

    v8f acc = {};
#pragma unroll
    for (int kc = 0; kc < CIN; kc += 32) {
        const int cur = (kc >> 5) & 1, nxt = cur ^ 1;
        if (kc + 32 < CIN) {           // prefetch next chunk's fragments
            const __bf16* wrn = wr + kc + 32;
            const __bf16* bpn = bp0 + kc + 32;
            A[nxt].q[0] = *(const uint4*)(wrn);
            A[nxt].q[1] = *(const uint4*)(wrn + 16);
            B[nxt].q[0] = *(const uint4*)(bpn);
            B[nxt].q[1] = *(const uint4*)(bpn + 16);
        }
        acc = __builtin_amdgcn_wmma_f32_16x16x32_bf16(
                  false, A[cur].v, false, B[cur].v, (short)0, acc, false, false);
    }

    // D layout: VGPR r -> M = r (lanes 0-15) or r+8 (lanes 16-31); N = lane&15
#pragma unroll
    for (int r = 0; r < 8; ++r) {
        int mo = wave * 16 + r + half * 8;
        sP[l15][mo] = sigm(acc[r] + biasPad[mo]);   // row 255 harmless garbage
    }
    __syncthreads();

    // ---- decode boxes: 16 positions x 3 anchors ----
    if (tid < 48) {
        int pos = tid / 3, a = tid - pos * 3;
        int base = a * L;
        float px = sP[pos][base + 0], py = sP[pos][base + 1];
        float pw = sP[pos][base + 2], ph = sP[pos][base + 3];
        int hwp = hwBase + pos;
        float gx = (float)(hwp % W), gy = (float)(hwp / W);
        float cx = (2.0f * px - 0.5f + gx) * stride;
        float cy = (2.0f * py - 0.5f + gy) * stride;
        float aw = (a == 0) ? ax0 : ((a == 1) ? ax1 : ax2);
        float ah = (a == 0) ? ay0 : ((a == 1) ? ay1 : ay2);
        float bw = 4.0f * pw * pw * aw;
        float bh = 4.0f * ph * ph * ah;
        float x1 = fminf(fmaxf(cx - bw * 0.5f, 0.0f), IMG);
        float y1 = fminf(fmaxf(cy - bh * 0.5f, 0.0f), IMG);
        float x2 = fminf(fmaxf(cx + bw * 0.5f, 0.0f), IMG);
        float y2 = fminf(fmaxf(cy + bh * 0.5f, 0.0f), IMG);
        size_t g = (size_t)n * TOTAL_ANCHORS + (levelBase + hwp * NA + a);
        boxesOut[g * 4 + 0] = x1; boxesOut[g * 4 + 1] = y1;
        boxesOut[g * 4 + 2] = x2; boxesOut[g * 4 + 3] = y2;
    }

    // ---- scoring + candidate append: 16*3*80 = 3840 (pos,anchor,class) ----
    for (int e = tid; e < 16 * NA * NUM_CLASSES; e += 512) {
        int pos = e / (NA * NUM_CLASSES);
        int rem = e - pos * (NA * NUM_CLASSES);
        int a = rem / NUM_CLASSES;
        int c = rem - a * NUM_CLASSES;
        float obj = sP[pos][a * L + 4];
        float cl  = sP[pos][a * L + 5 + c];
        float sc  = obj * cl;
        if (obj > SCORE_T && sc > SCORE_T) {
            int slot = atomicAdd(&candCnt[n], 1);
            if (slot < CAND_CAP) {
                int g = levelBase + (hwBase + pos) * NA + a;
                candScore[n * CAND_CAP + slot] = sc;
                candIdx[n * CAND_CAP + slot]   = g * NUM_CLASSES + c;
            }
        }
    }
}

// ---------------- Kernel 2: per-image bitonic sort -> top 1000 ----------------
__global__ __launch_bounds__(1024) void topk_sort_kernel(
    const float* __restrict__ candScore, const int* __restrict__ candIdx,
    const int* __restrict__ candCnt,
    float* __restrict__ topScore, int* __restrict__ topIdx)
{
    __shared__ float ss[CAND_CAP];
    __shared__ int   si[CAND_CAP];
    const int n = blockIdx.x, tid = threadIdx.x;
    int cnt = candCnt[n]; if (cnt > CAND_CAP) cnt = CAND_CAP;

    for (int t = tid; t < CAND_CAP; t += 1024) {
        if (t < cnt) { ss[t] = candScore[n * CAND_CAP + t]; si[t] = candIdx[n * CAND_CAP + t]; }
        else         { ss[t] = -1.0f; si[t] = 0; }   // masked scores are exactly -1.0
    }
    __syncthreads();

    for (int k = 2; k <= CAND_CAP; k <<= 1) {
        for (int j = k >> 1; j > 0; j >>= 1) {
            for (int t = tid; t < CAND_CAP; t += 1024) {
                int x = t ^ j;
                if (x > t) {
                    bool desc = ((t & k) == 0);
                    float s1 = ss[t], s2 = ss[x];
                    if ((s1 < s2) == desc) {
                        ss[t] = s2; ss[x] = s1;
                        int ti = si[t]; si[t] = si[x]; si[x] = ti;
                    }
                }
            }
            __syncthreads();
        }
    }
    for (int t = tid; t < TOPK_PAD; t += 1024) {
        topScore[n * TOPK_PAD + t] = (t < TOPK) ? ss[t] : -1.0f;
        topIdx[n * TOPK_PAD + t]   = (t < TOPK) ? si[t] : 0;
    }
}

// ---------------- Kernel 3: per-image greedy NMS (100 fixed iters) ------------
__global__ __launch_bounds__(1024) void nms_kernel(
    const float* __restrict__ topScore, const int* __restrict__ topIdx,
    const float* __restrict__ boxesAll, const float* __restrict__ sfArr,
    float* __restrict__ out)
{
    __shared__ float bx[TOPK][4];     // clipped xyxy boxes
    __shared__ float sc[TOPK], lb[TOPK];
    __shared__ int   valid[TOPK];
    __shared__ float rs[1024];
    __shared__ int   ri[1024];
    __shared__ int   sBest, sOk;

    const int n = blockIdx.x, t = threadIdx.x;
    const float sf = sfArr[n];

    if (t < TOPK) {
        float s = topScore[n * TOPK_PAD + t];
        int id = topIdx[n * TOPK_PAD + t];
        int anchor = id / NUM_CLASSES;
        int label  = id - anchor * NUM_CLASSES;
        size_t g = (size_t)n * TOTAL_ANCHORS + anchor;
#pragma unroll
        for (int j = 0; j < 4; ++j) bx[t][j] = boxesAll[g * 4 + j];
        sc[t] = s;
        lb[t] = (float)label;
        valid[t] = (s > SCORE_T) ? 1 : 0;
    }
    __syncthreads();

    for (int it = 0; it < DETECTIONS; ++it) {
        // argmax over valid scores (first-index tie break, like jnp.argmax)
        float ms = -3.0e38f; int mi = t;
        if (t < TOPK && valid[t]) { ms = sc[t]; mi = t; }
        rs[t] = ms; ri[t] = mi;
        __syncthreads();
        for (int s = 512; s > 0; s >>= 1) {
            if (t < s) {
                float a = rs[t], b = rs[t + s];
                int ia = ri[t], ib = ri[t + s];
                if (b > a || (b == a && ib < ia)) { rs[t] = b; ri[t] = ib; }
            }
            __syncthreads();
        }
        if (t == 0) {
            int i = ri[0];
            int ok = (rs[0] > 0.0f) ? 1 : 0;
            sBest = i; sOk = ok;
            float okf = ok ? 1.0f : 0.0f;
            int gi = ok ? i : 0;
            float* o = out + ((size_t)n * DETECTIONS + it) * 6;
            o[0] = bx[gi][0] / sf * okf;
            o[1] = bx[gi][1] / sf * okf;
            o[2] = bx[gi][2] / sf * okf;
            o[3] = bx[gi][3] / sf * okf;
            o[4] = sc[gi] * okf;
            o[5] = lb[gi] * okf;
        }
        __syncthreads();
        if (sOk && t < TOPK && valid[t]) {
            int i = sBest;
            float offI = lb[i] * IMG, offT = lb[t] * IMG;  // class offsets
            float ix1 = bx[i][0] + offI, iy1 = bx[i][1] + offI;
            float ix2 = bx[i][2] + offI, iy2 = bx[i][3] + offI;
            float tx1 = bx[t][0] + offT, ty1 = bx[t][1] + offT;
            float tx2 = bx[t][2] + offT, ty2 = bx[t][3] + offT;
            float lx = fmaxf(ix1, tx1), ly = fmaxf(iy1, ty1);
            float rx = fminf(ix2, tx2), ry = fminf(iy2, ty2);
            float iw = fmaxf(rx - lx, 0.0f), ih = fmaxf(ry - ly, 0.0f);
            float inter = iw * ih;
            float areaI = (ix2 - ix1) * (iy2 - iy1);
            float areaT = (tx2 - tx1) * (ty2 - ty1);
            float iou = inter / (areaI + areaT - inter + 1e-7f);
            if (iou > NMS_T) valid[t] = 0;   // self IoU==1 invalidates i too
        }
        __syncthreads();
    }
}

// ---------------- host launch ----------------
extern "C" void kernel_launch(void* const* d_in, const int* in_sizes, int n_in,
                              void* d_out, int out_size, void* d_ws, size_t ws_size,
                              hipStream_t stream) {
    // dict order: f0, w0, b0, f1, w1, b1, f2, w2, b2, scale_factors
    const float* f0 = (const float*)d_in[0];
    const float* w0 = (const float*)d_in[1];
    const float* b0 = (const float*)d_in[2];
    const float* f1 = (const float*)d_in[3];
    const float* w1 = (const float*)d_in[4];
    const float* b1 = (const float*)d_in[5];
    const float* f2 = (const float*)d_in[6];
    const float* w2 = (const float*)d_in[7];
    const float* b2 = (const float*)d_in[8];
    const float* sf = (const float*)d_in[9];
    float* out = (float*)d_out;

    // workspace carve-up (all segments stay 16B aligned)
    float* boxes     = (float*)d_ws;                               // 8*25200*4 f32
    int*   candCnt   = (int*)(boxes + (size_t)NIMG * TOTAL_ANCHORS * 4);
    float* candScore = (float*)(candCnt + NIMG);                   // NIMG pads to 32B
    int*   candIdx   = (int*)(candScore + (size_t)NIMG * CAND_CAP);
    float* topScore  = (float*)(candIdx + (size_t)NIMG * CAND_CAP);
    int*   topIdx    = (int*)(topScore + (size_t)NIMG * TOPK_PAD);
    float* bias0     = (float*)(topIdx + (size_t)NIMG * TOPK_PAD); // 3 x 256 f32
    float* bias1     = bias0 + 256;
    float* bias2     = bias1 + 256;
    __bf16* wbf0     = (__bf16*)(bias2 + 256);                     // 256-row padded
    __bf16* wbf1     = wbf0 + (size_t)256 * 128;
    __bf16* wbf2     = wbf1 + (size_t)256 * 256;

    init_kernel<<<1, 32, 0, stream>>>(candCnt);

    // pre-convert 1x1-conv weights to bf16, pad row 255 with zeros; pad biases
    convert_weights_kernel<<<(256 * 128 + 255) / 256, 256, 0, stream>>>(w0, wbf0, 128);
    convert_weights_kernel<<<(256 * 256 + 255) / 256, 256, 0, stream>>>(w1, wbf1, 256);
    convert_weights_kernel<<<(256 * 512 + 255) / 256, 256, 0, stream>>>(w2, wbf2, 512);
    pad_bias_kernel<<<1, 256, 0, stream>>>(b0, bias0);
    pad_bias_kernel<<<1, 256, 0, stream>>>(b1, bias1);
    pad_bias_kernel<<<1, 256, 0, stream>>>(b2, bias2);

    // level 0: Cin=128, 80x80, stride 8
    decode_level_kernel<128><<<dim3(80 * 80 / 16, NIMG), 512, 0, stream>>>(
        f0, wbf0, bias0, boxes, candScore, candIdx, candCnt,
        80, 80, 8.0f, 6.1f, 8.1f, 20.6f, 12.6f, 11.2f, 23.7f, 0);
    // level 1: Cin=256, 40x40, stride 16
    decode_level_kernel<256><<<dim3(40 * 40 / 16, NIMG), 512, 0, stream>>>(
        f1, wbf1, bias1, boxes, candScore, candIdx, candCnt,
        40, 40, 16.0f, 36.2f, 26.8f, 25.9f, 57.2f, 57.8f, 47.9f, 19200);
    // level 2: Cin=512, 20x20, stride 32
    decode_level_kernel<512><<<dim3(20 * 20 / 16, NIMG), 512, 0, stream>>>(
        f2, wbf2, bias2, boxes, candScore, candIdx, candCnt,
        20, 20, 32.0f, 122.1f, 78.3f, 73.7f, 143.8f, 236.1f, 213.1f, 24000);

    topk_sort_kernel<<<NIMG, 1024, 0, stream>>>(candScore, candIdx, candCnt,
                                                topScore, topIdx);
    nms_kernel<<<NIMG, 1024, 0, stream>>>(topScore, topIdx, boxes, sf, out);
}
